// Attention_spatial_2078764172258
// MI455X (gfx1250) — matrix operand order repaired and checked by reading, not compiled, not run on hardware
//
#include <hip/hip_runtime.h>
#include <hip/hip_bf16.h>

// ---------------- problem constants ----------------
#define BB     4
#define CC     256
#define HH     48
#define WWID   48
#define SS     (HH * WWID)        // 2304
#define NHEAD  4
#define HD     64                 // head dim
#define GROUPS 16
#define CPG    (CC / GROUPS)      // 16
#define EPS    1e-5f
#define SCALE  0.0625f            // 1/sqrt(256)

typedef __attribute__((ext_vector_type(16))) _Float16 v16h;
typedef __attribute__((ext_vector_type(8)))  float    v8f;

union AFrag {
    v16h v;
    _Float16 h[16];
    uint4 u[2];
};

__device__ inline v8f wmma_f16(v16h a, v16h b, v8f c) {
    // D = A(16x32 f16) * B(32x16 f16) + C(16x16 f32)
    return __builtin_amdgcn_wmma_f32_16x16x32_f16(
        /*neg_a=*/false, a, /*neg_b=*/false, b,
        /*c_mod=*/(short)0, c, /*reuse_a=*/false, /*reuse_b=*/false);
}

// ---------------- kernel 1: weight f32 -> f16 ----------------
__global__ void cvt_weights_kernel(const float* __restrict__ qkvw,
                                   const float* __restrict__ outw,
                                   _Float16* __restrict__ qkvw_h,
                                   _Float16* __restrict__ outw_h) {
    int i = blockIdx.x * blockDim.x + threadIdx.x;
    if (i < 3 * CC * CC) qkvw_h[i] = (_Float16)qkvw[i];
    if (i < CC * CC)     outw_h[i] = (_Float16)outw[i];
}

// ---------------- kernel 2: GroupNorm -> normT[b][s][c] (f16) ----------------
__global__ void groupnorm_kernel(const float* __restrict__ x,
                                 const float* __restrict__ gw,
                                 const float* __restrict__ gb,
                                 _Float16* __restrict__ normT) {
    const int bg = blockIdx.x;            // b * GROUPS + g
    const int b  = bg / GROUPS;
    const int g  = bg % GROUPS;
    const int NE = CPG * SS;              // 36864 elements per (b, group)
    const float* xp = x + ((size_t)b * CC + (size_t)g * CPG) * SS;

    float sum = 0.f, sq = 0.f;
    for (int i = threadIdx.x; i < NE; i += blockDim.x) {
        float v = xp[i];
        sum += v;
        sq  += v * v;
    }
    __shared__ float s_sum[256];
    __shared__ float s_sq[256];
    int tid = threadIdx.x;
    s_sum[tid] = sum;
    s_sq[tid]  = sq;
    __syncthreads();
    for (int st = 128; st > 0; st >>= 1) {
        if (tid < st) {
            s_sum[tid] += s_sum[tid + st];
            s_sq[tid]  += s_sq[tid + st];
        }
        __syncthreads();
    }
    const float mean = s_sum[0] / (float)NE;
    const float var  = s_sq[0] / (float)NE - mean * mean;
    const float inv  = rsqrtf(var + EPS);

    // norm = x*(inv*w) + (b - mean*inv*w): no per-element division/modulo
    for (int cl = 0; cl < CPG; cl++) {
        const int c = g * CPG + cl;
        const float wc = gw[c] * inv;
        const float bc = gb[c] - mean * wc;
        const float* xr = xp + (size_t)cl * SS;
        for (int s = threadIdx.x; s < SS; s += blockDim.x) {
            normT[((size_t)b * SS + s) * CC + c] = (_Float16)(xr[s] * wc + bc);
        }
    }
}

// ---------------- kernel 3: QKV GEMM (one wave = 16x64 tile, SW-pipelined) ----
// qkv[b, o, s] = sum_c qkvw[o,c] * norm[b,c,s]
// A = qkvw_h [768][256] row-major, B = normT [b][s][256] (c contiguous).
// Scatter: q,k stored [bn][s][64]; v stored transposed [bn][d][s].
__global__ void qkv_gemm_kernel(const _Float16* __restrict__ Wh,
                                const _Float16* __restrict__ normT,
                                _Float16* __restrict__ q,
                                _Float16* __restrict__ k,
                                _Float16* __restrict__ vT) {
    const int wid  = blockIdx.x * (blockDim.x >> 5) + (threadIdx.x >> 5);
    const int NT64 = SS / 64;             // 36
    const int tn   = wid % NT64;
    int t2         = wid / NT64;
    const int tm   = t2 % 48;             // 768/16
    const int b    = t2 / 48;
    const int lane = threadIdx.x & 31;
    const int ln   = lane & 15;
    const int gr   = lane >> 4;

    // wave-uniform output-channel decomposition (16-row tiles never cross
    // a q/k/v boundary since those are 64-aligned)
    const int obase = tm * 16;
    const int head  = obase / 192;
    const int rr    = obase % 192;
    const int sel   = rr / 64;            // 0=q, 1=k, 2=v
    const int dbase = rr % 64;

    const _Float16* arow = Wh + ((size_t)(obase + ln)) * CC + gr * 16;
    const _Float16* brow = normT + ((size_t)b * SS + tn * 64 + ln) * CC + gr * 16;

    v8f acc[4];
#pragma unroll
    for (int nt = 0; nt < 4; nt++) acc[nt] = (v8f){};

    // double-buffered fragments: issue k-step kk+1 loads before kk's WMMAs
    AFrag a[2], bf[2][4];
    auto load_step = [&](int kk, int buf) {
        const _Float16* ap = arow + kk * 32;
        a[buf].u[0] = *(const uint4*)ap;
        a[buf].u[1] = *(const uint4*)(ap + 8);
#pragma unroll
        for (int nt = 0; nt < 4; nt++) {
            const _Float16* bp = brow + (size_t)nt * 16 * CC + kk * 32;
            bf[buf][nt].u[0] = *(const uint4*)bp;
            bf[buf][nt].u[1] = *(const uint4*)(bp + 8);
        }
    };
    load_step(0, 0);
#pragma unroll
    for (int kk = 0; kk < 8; kk++) {      // K = 256 = 8 * 32
        const int cur = kk & 1;
        if (kk < 7) load_step(kk + 1, cur ^ 1);
#pragma unroll
        for (int nt = 0; nt < 4; nt++)
            acc[nt] = wmma_f16(a[cur].v, bf[cur][nt].v, acc[nt]);
    }

    const int bn = b * NHEAD + head;
#pragma unroll
    for (int nt = 0; nt < 4; nt++) {
#pragma unroll
        for (int j = 0; j < 8; j++) {
            const int d = dbase + j + 8 * gr;
            const int s = tn * 64 + nt * 16 + ln;
            const _Float16 val = (_Float16)acc[nt][j];
            if (sel == 0)      q [((size_t)bn * SS + s) * HD + d] = val;
            else if (sel == 1) k [((size_t)bn * SS + s) * HD + d] = val;
            else               vT[((size_t)bn * HD + d) * SS + s] = val;
        }
    }
}

// ---------------- kernel 4: flash attention per (b, head) ----------------
// Q,K stored [bn][s][64]; V stored [bn][d][s]. One wave per 16-query tile.
__global__ void attn_kernel(const _Float16* __restrict__ q,
                            const _Float16* __restrict__ k,
                            const _Float16* __restrict__ vT,
                            _Float16* __restrict__ attnT) {
    __shared__ _Float16 plds[8][16 * 32];
    const int wslot = threadIdx.x >> 5;
    const int wid   = blockIdx.x * 8 + wslot;      // 0..2303
    const int QT    = SS / 16;                     // 144
    const int qt    = wid % QT;
    const int bn    = wid / QT;                    // b*NHEAD + n
    const int lane  = threadIdx.x & 31;
    const int ln    = lane & 15;
    const int gr    = lane >> 4;

    const _Float16* Q  = q  + (size_t)bn * SS * HD;
    const _Float16* K  = k  + (size_t)bn * SS * HD;
    const _Float16* Vt = vT + (size_t)bn * HD * SS;

    // Q A-fragments (kept in registers): rows = queries, K-dim = d
    AFrag aq0, aq1;
    {
        const _Float16* qp = Q + (size_t)(qt * 16 + ln) * HD + gr * 16;
        aq0.u[0] = *(const uint4*)qp;
        aq0.u[1] = *(const uint4*)(qp + 8);
        aq1.u[0] = *(const uint4*)(qp + 32);
        aq1.u[1] = *(const uint4*)(qp + 40);
    }

    float m_run[8], l_run[8];
    v8f o_acc[4];
#pragma unroll
    for (int j = 0; j < 8; j++) { m_run[j] = -1e30f; l_run[j] = 0.f; }
#pragma unroll
    for (int t = 0; t < 4; t++) o_acc[t] = (v8f){};

    // double-buffered K fragments across 32-key iterations
    AFrag bk[2][4];
    auto load_k = [&](int kb, int buf) {
        const _Float16* kp0 = K + (size_t)(kb + ln) * HD + gr * 16;
        const _Float16* kp1 = K + (size_t)(kb + 16 + ln) * HD + gr * 16;
        bk[buf][0].u[0] = *(const uint4*)kp0;          // keys kb..+15, d 0..31
        bk[buf][0].u[1] = *(const uint4*)(kp0 + 8);
        bk[buf][1].u[0] = *(const uint4*)(kp0 + 32);   // keys kb..+15, d 32..63
        bk[buf][1].u[1] = *(const uint4*)(kp0 + 40);
        bk[buf][2].u[0] = *(const uint4*)kp1;          // keys kb+16..+31, d 0..31
        bk[buf][2].u[1] = *(const uint4*)(kp1 + 8);
        bk[buf][3].u[0] = *(const uint4*)(kp1 + 32);   // keys kb+16..+31, d 32..63
        bk[buf][3].u[1] = *(const uint4*)(kp1 + 40);
    };
    load_k(0, 0);

    for (int kb = 0, cur = 0; kb < SS; kb += 32, cur ^= 1) {
        // ---- V fragments for this block: independent, issue first
        AFrag bv[4];
#pragma unroll
        for (int t = 0; t < 4; t++) {
            const _Float16* vp = Vt + (size_t)(t * 16 + ln) * SS + kb + gr * 16;
            bv[t].u[0] = *(const uint4*)vp;
            bv[t].u[1] = *(const uint4*)(vp + 8);
        }
        // ---- prefetch next V block one iteration ahead (global_prefetch_b8)
        if (kb + 32 < SS) {
            __builtin_prefetch((const void*)(Vt + (size_t)lane * SS + kb + 32), 0, 3);
            __builtin_prefetch((const void*)(Vt + (size_t)(lane + 32) * SS + kb + 32), 0, 3);
        }

        // ---- scores for 32 keys with current K buffer
        v8f sc[2];
        {
            v8f c0 = {}, c1 = {};
            c0 = wmma_f16(aq0.v, bk[cur][0].v, c0);
            c1 = wmma_f16(aq0.v, bk[cur][2].v, c1);
            c0 = wmma_f16(aq1.v, bk[cur][1].v, c0);
            c1 = wmma_f16(aq1.v, bk[cur][3].v, c1);
            sc[0] = c0 * SCALE;
            sc[1] = c1 * SCALE;
        }
        // ---- issue next K block's loads; softmax VALU below hides them
        if (kb + 32 < SS) load_k(kb + 32, cur ^ 1);

        // ---- online softmax (row stats live on C/D layout: row = j + 8*gr)
        float corr[8];
#pragma unroll
        for (int j = 0; j < 8; j++) {
            float t = fmaxf(sc[0][j], sc[1][j]);
#pragma unroll
            for (int off = 1; off < 16; off <<= 1)
                t = fmaxf(t, __shfl_xor(t, off, 32));
            float mnew = fmaxf(m_run[j], t);
            corr[j] = __expf(m_run[j] - mnew);
            float p0 = __expf(sc[0][j] - mnew);
            float p1 = __expf(sc[1][j] - mnew);
            sc[0][j] = p0;
            sc[1][j] = p1;
            float ps = p0 + p1;
#pragma unroll
            for (int off = 1; off < 16; off <<= 1)
                ps += __shfl_xor(ps, off, 32);
            l_run[j] = l_run[j] * corr[j] + ps;
            m_run[j] = mnew;
        }
#pragma unroll
        for (int t = 0; t < 4; t++)
#pragma unroll
            for (int j = 0; j < 8; j++) o_acc[t][j] *= corr[j];

        // ---- stage P (16x32 f16) through per-wave LDS to form A-fragment
        _Float16* pl = &plds[wslot][0];
#pragma unroll
        for (int j = 0; j < 8; j++) {
            int m = j + 8 * gr;
            pl[m * 32 + ln]      = (_Float16)sc[0][j];
            pl[m * 32 + 16 + ln] = (_Float16)sc[1][j];
        }
        asm volatile("s_wait_dscnt 0x0" ::: "memory");  // DS in-order per wave
        AFrag ap;
        const _Float16* pp = pl + ln * 32 + gr * 16;
        ap.u[0] = *(const uint4*)pp;
        ap.u[1] = *(const uint4*)(pp + 8);

        // ---- O += P(16x32) * V^T(32x16): V fragments already in flight
#pragma unroll
        for (int t = 0; t < 4; t++)
            o_acc[t] = wmma_f16(ap.v, bv[t].v, o_acc[t]);
    }

    // ---- epilogue: normalize and store transposed [b][s][c]
    const int b = bn >> 2;
    const int n = bn & 3;
#pragma unroll
    for (int j = 0; j < 8; j++) {
        float inv = 1.0f / l_run[j];
        int s = qt * 16 + j + 8 * gr;
#pragma unroll
        for (int t = 0; t < 4; t++) {
            attnT[((size_t)b * SS + s) * CC + n * 64 + t * 16 + ln] =
                (_Float16)(o_acc[t][j] * inv);
        }
    }
}

// ---------------- kernel 5: projection + bias + residual (16x64/wave) --------
__global__ void proj_kernel(const _Float16* __restrict__ Wh,
                            const _Float16* __restrict__ attnT,
                            const float* __restrict__ ob,
                            const float* __restrict__ x,
                            float* __restrict__ out) {
    const int wid  = blockIdx.x * (blockDim.x >> 5) + (threadIdx.x >> 5);
    const int NT64 = SS / 64;             // 36
    const int tn   = wid % NT64;
    int t2         = wid / NT64;
    const int tm   = t2 % 16;             // 256/16
    const int b    = t2 / 16;
    const int lane = threadIdx.x & 31;
    const int ln   = lane & 15;
    const int gr   = lane >> 4;

    const _Float16* arow = Wh + ((size_t)(tm * 16 + ln)) * CC + gr * 16;
    const _Float16* brow = attnT + ((size_t)b * SS + tn * 64 + ln) * CC + gr * 16;

    v8f acc[4];
#pragma unroll
    for (int nt = 0; nt < 4; nt++) acc[nt] = (v8f){};

    AFrag a[2], bf[2][4];
    auto load_step = [&](int kk, int buf) {
        const _Float16* ap = arow + kk * 32;
        a[buf].u[0] = *(const uint4*)ap;
        a[buf].u[1] = *(const uint4*)(ap + 8);
#pragma unroll
        for (int nt = 0; nt < 4; nt++) {
            const _Float16* bp = brow + (size_t)nt * 16 * CC + kk * 32;
            bf[buf][nt].u[0] = *(const uint4*)bp;
            bf[buf][nt].u[1] = *(const uint4*)(bp + 8);
        }
    };
    load_step(0, 0);
#pragma unroll
    for (int kk = 0; kk < 8; kk++) {
        const int cur = kk & 1;
        if (kk < 7) load_step(kk + 1, cur ^ 1);
#pragma unroll
        for (int nt = 0; nt < 4; nt++)
            acc[nt] = wmma_f16(a[cur].v, bf[cur][nt].v, acc[nt]);
    }

#pragma unroll
    for (int nt = 0; nt < 4; nt++) {
#pragma unroll
        for (int j = 0; j < 8; j++) {
            const int o = tm * 16 + j + 8 * gr;
            const int s = tn * 64 + nt * 16 + ln;
            const size_t idx = ((size_t)b * CC + o) * SS + s;
            out[idx] = acc[nt][j] + ob[o] + x[idx];
        }
    }
}

// ---------------- launcher ----------------
extern "C" void kernel_launch(void* const* d_in, const int* in_sizes, int n_in,
                              void* d_out, int out_size, void* d_ws, size_t ws_size,
                              hipStream_t stream) {
    const float* x    = (const float*)d_in[0];
    const float* gw   = (const float*)d_in[1];
    const float* gb   = (const float*)d_in[2];
    const float* qkvw = (const float*)d_in[3];
    const float* outw = (const float*)d_in[4];
    const float* outb = (const float*)d_in[5];
    float* out = (float*)d_out;

    // workspace layout (f16 tensors), ~24.1 MB total
    char* ws = (char*)d_ws;
    size_t off = 0;
    _Float16* normT   = (_Float16*)(ws + off); off += (size_t)BB * SS * CC * 2;          // 4.7 MB
    _Float16* qkvw_h  = (_Float16*)(ws + off); off += (size_t)3 * CC * CC * 2;           // 384 KB
    _Float16* outw_h  = (_Float16*)(ws + off); off += (size_t)CC * CC * 2;               // 128 KB
    _Float16* qh      = (_Float16*)(ws + off); off += (size_t)BB * NHEAD * SS * HD * 2;  // 4.7 MB
    _Float16* kh      = (_Float16*)(ws + off); off += (size_t)BB * NHEAD * SS * HD * 2;
    _Float16* vTh     = (_Float16*)(ws + off); off += (size_t)BB * NHEAD * HD * SS * 2;
    _Float16* attnT   = (_Float16*)(ws + off); off += (size_t)BB * SS * CC * 2;

    // 1) weights -> f16
    {
        int n = 3 * CC * CC;
        cvt_weights_kernel<<<(n + 255) / 256, 256, 0, stream>>>(qkvw, outw, qkvw_h, outw_h);
    }
    // 2) GroupNorm -> normT
    groupnorm_kernel<<<BB * GROUPS, 256, 0, stream>>>(x, gw, gb, normT);
    // 3) QKV GEMM: 4 batches * 48 (M16) * 36 (N64) wave-tiles, 4 waves/block
    qkv_gemm_kernel<<<(BB * 48 * (SS / 64)) / 4, 128, 0, stream>>>(qkvw_h, normT, qh, kh, vTh);
    // 4) attention: 16 (b,n) * 144 query tiles = 2304 waves, 8 waves/block
    attn_kernel<<<(BB * NHEAD * (SS / 16)) / 8, 256, 0, stream>>>(qh, kh, vTh, attnT);
    // 5) projection + bias + residual: 4 * 16 (M16) * 36 (N64) waves
    proj_kernel<<<(BB * 16 * (SS / 64)) / 4, 128, 0, stream>>>(outw_h, attnT, outb, x, out);
}
